// Phi_10282151707316
// MI455X (gfx1250) — matrix operand order repaired and checked by reading, not compiled, not run on hardware
//
#include <hip/hip_runtime.h>
#include <hip/hip_bf16.h>
#include <math.h>

#define NSEG 8192
#define CIN 128
#define COUT 256
#define TM 16
#define LN_EPS 1e-5f
#define XPAD 136   // bf16 row stride for staged xm tile in lam_gemm
#define FSTR 132   // f32 row stride for TDM-padded x tile (128 + 4 pad DWORDs)
#define HPAD 260   // f32 row stride for h tile in LDS

typedef __attribute__((ext_vector_type(16))) __bf16 v16bf;
typedef __attribute__((ext_vector_type(8)))  float  v8f;
typedef __attribute__((ext_vector_type(4)))  unsigned int v4u;
typedef __attribute__((ext_vector_type(8)))  int v8i;
typedef __attribute__((ext_vector_type(4)))  int v4i;

// ---------------- LDS byte-offset of a shared-memory pointer ----------------
__device__ __forceinline__ unsigned lds_addr_of(const void* p) {
    return (unsigned)(unsigned long long)(__attribute__((address_space(3))) const char*)p;
}

// ---------------- TDM: DMA a 16x128 f32 tile (row-padded +16B) into LDS -----
// D# per CDNA5 ISA ch.8: group0 = {flags, lds_addr, global_addr, type=2},
// group1 = {data_size=4B | pad(128 DW interval, 4 DW amount), tensor_dim0=128,
//           tensor_dim1=rowsRemain (OOB rows read as 0), tile 128x16, stride0=128}
__device__ __forceinline__ void tdm_load_tile(const float* gsrc, unsigned ldsaddr,
                                              unsigned rowsRemain) {
    unsigned long long ga = (unsigned long long)gsrc;
    v4u g0;
    g0[0] = 1u;                                          // count=1, user descriptor
    g0[1] = ldsaddr;                                     // LDS byte address
    g0[2] = (unsigned)(ga & 0xFFFFFFFFull);              // global_addr[31:0]
    g0[3] = (unsigned)((ga >> 32) & 0x1FFFFFFu) | (2u << 30); // addr[56:32] | type=2
    v8i g1;
    g1[0] = (int)((2u << 16) | (1u << 20) | (6u << 22) | (3u << 25)); // 4B, pad en, 128DW/4DW
    g1[1] = (int)(128u << 16);                           // tensor_dim0[15:0]=128
    g1[2] = (int)((rowsRemain & 0xFFFFu) << 16);         // dim0 hi=0 | tensor_dim1 lo
    g1[3] = (int)(((rowsRemain >> 16) & 0xFFFFu) | (128u << 16)); // dim1 hi | tile_dim0=128
    g1[4] = (int)(TM);                                   // tile_dim1=16, tile_dim2=0
    g1[5] = (int)(128u);                                 // tensor_dim0_stride lo32 = 128
    g1[6] = 0;
    g1[7] = 0;
    v4i gz4 = {0, 0, 0, 0};
    v8i gz8 = {0, 0, 0, 0, 0, 0, 0, 0};
    __builtin_amdgcn_tensor_load_to_lds(g0, g1, gz4, gz4, gz8, 0);
}

// ---------------- workspace fill: xm = -inf ----------------
__global__ void fill_neginf(float* __restrict__ p, int n) {
    int i = blockIdx.x * blockDim.x + threadIdx.x;
    if (i < n) p[i] = -INFINITY;
}

// ---------------- weights -> bf16 hi/lo pairs ----------------
__global__ void cvt_weights(const float* __restrict__ gw, const float* __restrict__ lw,
                            __bf16* __restrict__ gwh, __bf16* __restrict__ gwl,
                            __bf16* __restrict__ lwh, __bf16* __restrict__ lwl) {
    int i = blockIdx.x * blockDim.x + threadIdx.x;
    if (i >= COUT * CIN) return;
    float g = gw[i];
    __bf16 gh = (__bf16)g;
    gwh[i] = gh;
    gwl[i] = (__bf16)(g - (float)gh);
    float l = lw[i];
    __bf16 lh = (__bf16)l;
    lwh[i] = lh;
    lwl[i] = (__bf16)(l - (float)lh);
}

// ---------------- segment max (batch ids sorted) ----------------
__global__ void seg_max(const float* __restrict__ x, const int* __restrict__ batch,
                        float* __restrict__ xm, int N) {
    const int ROWS = 256;
    int r0 = blockIdx.x * ROWS;
    if (r0 >= N) return;
    int c = threadIdx.x;
    int rend = min(r0 + ROWS, N);
    float m = -INFINITY;
    int seg = batch[r0];
    for (int r = r0; r < rend; ++r) {
        int s = batch[r];
        if (s != seg) {
            atomicMax(&xm[(long)seg * CIN + c], m);
            m = -INFINITY;
            seg = s;
        }
        m = fmaxf(m, x[(long)r * CIN + c]);
    }
    atomicMax(&xm[(long)seg * CIN + c], m);
}

// ---------------- A fragment from bf16 LDS tile (lam_gemm path) -------------
__device__ __forceinline__ v16bf load_afrag(const __bf16* __restrict__ s, int lane, int ks) {
    int m = lane & 15;
    int kb = ks * 32 + ((lane < 16) ? 0 : 16);
    const __bf16* p = s + m * XPAD + kb;
    v16bf a;
#pragma unroll
    for (int i = 0; i < 8; ++i) {
        a[i]     = p[i];
        a[i + 8] = p[8 + i];
    }
    return a;
}

// ---------------- A hi/lo fragments straight from f32 LDS tile --------------
__device__ __forceinline__ void load_afrag_f32(const float* __restrict__ s, int lane, int ks,
                                               v16bf& ah, v16bf& al) {
    int m = lane & 15;
    int kb = ks * 32 + ((lane < 16) ? 0 : 16);
    const float* p = s + m * FSTR + kb;
#pragma unroll
    for (int i = 0; i < 8; ++i) {
        float v0 = p[i], v1 = p[8 + i];
        __bf16 h0 = (__bf16)v0, h1 = (__bf16)v1;
        ah[i] = h0;
        ah[i + 8] = h1;
        al[i] = (__bf16)(v0 - (float)h0);
        al[i + 8] = (__bf16)(v1 - (float)h1);
    }
}

// ---------------- lambda projection: lam = fix(xm) @ lambda_w^T -------------
__global__ void __launch_bounds__(512, 1)
lam_gemm(const float* __restrict__ xm,
         const __bf16* __restrict__ lwh, const __bf16* __restrict__ lwl,
         float* __restrict__ lam) {
    __shared__ __bf16 sxh[TM * XPAD];
    __shared__ __bf16 sxl[TM * XPAD];
    int tid = threadIdx.x, wave = tid >> 5, lane = tid & 31;
    int nCol = wave * 16 + (lane & 15);
    int kOff = (lane < 16) ? 0 : 16;

    v16bf bh[4], bl[4];
#pragma unroll
    for (int ks = 0; ks < 4; ++ks) {
        bh[ks] = *(const v16bf*)(lwh + nCol * CIN + ks * 32 + kOff);
        bl[ks] = *(const v16bf*)(lwl + nCol * CIN + ks * 32 + kOff);
    }

    int r0 = blockIdx.x * TM;
    {
        int row = tid >> 5;
        float4 xv = *(const float4*)(xm + (long)(r0 + row) * CIN + lane * 4);
        float vals[4] = {xv.x, xv.y, xv.z, xv.w};
        __bf16* ph = sxh + row * XPAD + lane * 4;
        __bf16* pl = sxl + row * XPAD + lane * 4;
#pragma unroll
        for (int j = 0; j < 4; ++j) {
            float v = vals[j];
            v = (v < -3.0e38f) ? 0.0f : v;   // empty segment (-inf) -> 0
            __bf16 h = (__bf16)v;
            ph[j] = h;
            pl[j] = (__bf16)(v - (float)h);
        }
    }
    __syncthreads();

    v8f c = {0.f, 0.f, 0.f, 0.f, 0.f, 0.f, 0.f, 0.f};
#pragma unroll
    for (int ks = 0; ks < 4; ++ks) {
        v16bf ah = load_afrag(sxh, lane, ks);
        v16bf al = load_afrag(sxl, lane, ks);
        c = __builtin_amdgcn_wmma_f32_16x16x32_bf16(false, ah, false, bh[ks], (short)0, c, false, false);
        c = __builtin_amdgcn_wmma_f32_16x16x32_bf16(false, ah, false, bl[ks], (short)0, c, false, false);
        c = __builtin_amdgcn_wmma_f32_16x16x32_bf16(false, al, false, bh[ks], (short)0, c, false, false);
    }

    int col = wave * 16 + (lane & 15);
    int mBase = (lane < 16) ? 0 : 8;
#pragma unroll
    for (int v = 0; v < 8; ++v) {
        int m = v + mBase;
        lam[(long)(r0 + m) * COUT + col] = c[v];
    }
}

// ------- fused main: TDM-double-buffered GEMM(bf16x3) - lam[batch] + LN + PReLU
__global__ void __launch_bounds__(512, 1)
main_fused(const float* __restrict__ x, const int* __restrict__ batch,
           const __bf16* __restrict__ gwh, const __bf16* __restrict__ gwl,
           const float* __restrict__ gb, const float* __restrict__ lam,
           const float* __restrict__ lnw, const float* __restrict__ lnb,
           const float* __restrict__ pa, float* __restrict__ out, int N) {
    __shared__ float sxf[2][TM * FSTR + 4];   // TDM-padded f32 x tiles (double buffer)
    __shared__ float sH[TM * HPAD];
    __shared__ int   sSeg[TM];

    int tid = threadIdx.x, wave = tid >> 5, lane = tid & 31;
    int nCol = wave * 16 + (lane & 15);
    int kOff = (lane < 16) ? 0 : 16;

    // persistent B panel: this wave's 16 output columns, all K, hi+lo (64 VGPRs)
    v16bf bh[4], bl[4];
#pragma unroll
    for (int ks = 0; ks < 4; ++ks) {
        bh[ks] = *(const v16bf*)(gwh + nCol * CIN + ks * 32 + kOff);
        bl[ks] = *(const v16bf*)(gwl + nCol * CIN + ks * 32 + kOff);
    }
    float slope = pa[0];
    int nTiles = (N + TM - 1) / TM;

    unsigned ldsBuf0 = lds_addr_of(&sxf[0][0]);
    unsigned ldsBuf1 = lds_addr_of(&sxf[1][0]);

    // prologue: DMA first tile into buffer 0 (wave 0 owns the TDM)
    int t0 = blockIdx.x;
    if (wave == 0 && t0 < nTiles)
        tdm_load_tile(x + (size_t)t0 * TM * CIN, ldsBuf0, (unsigned)(N - t0 * TM));

    int iter = 0;
    for (int t = t0; t < nTiles; t += gridDim.x, ++iter) {
        int cur = iter & 1;
        int r0 = t * TM;
        int tn = t + gridDim.x;

        // issue DMA for next tile into the other buffer, then wait for current
        if (wave == 0) {
            if (tn < nTiles) {
                tdm_load_tile(x + (size_t)tn * TM * CIN, cur ? ldsBuf0 : ldsBuf1,
                              (unsigned)(N - tn * TM));
                __builtin_amdgcn_s_wait_tensorcnt(1);
            } else {
                __builtin_amdgcn_s_wait_tensorcnt(0);
            }
        }
        if (tid < TM) sSeg[tid] = (r0 + tid < N) ? batch[r0 + tid] : 0;
        if (tid == 0 && tn < nTiles) __builtin_prefetch(batch + tn * TM, 0, 0);
        __syncthreads();

        const float* sx = &sxf[cur][0];
        v8f c = {0.f, 0.f, 0.f, 0.f, 0.f, 0.f, 0.f, 0.f};
#pragma unroll
        for (int ks = 0; ks < 4; ++ks) {
            v16bf ah, al;
            load_afrag_f32(sx, lane, ks, ah, al);
            c = __builtin_amdgcn_wmma_f32_16x16x32_bf16(false, ah, false, bh[ks], (short)0, c, false, false);
            c = __builtin_amdgcn_wmma_f32_16x16x32_bf16(false, ah, false, bl[ks], (short)0, c, false, false);
            c = __builtin_amdgcn_wmma_f32_16x16x32_bf16(false, al, false, bh[ks], (short)0, c, false, false);
        }

        {   // epilogue: + gamma_b - lam[batch] -> LDS h tile
            int col = wave * 16 + (lane & 15);
            float gbv = gb[col];
            int mBase = (lane < 16) ? 0 : 8;
#pragma unroll
            for (int v = 0; v < 8; ++v) {
                int m = v + mBase;
                sH[m * HPAD + col] = c[v] + gbv - lam[(long)sSeg[m] * COUT + col];
            }
        }
        __syncthreads();

        {   // LayerNorm + PReLU: wave w <-> row w; lane handles 8 cols
            int m = wave;
            int r = r0 + m;
            const float* hp = sH + m * HPAD + lane * 8;
            float4 h0 = *(const float4*)(hp);
            float4 h1 = *(const float4*)(hp + 4);
            float s = h0.x + h0.y + h0.z + h0.w + h1.x + h1.y + h1.z + h1.w;
            float q = h0.x * h0.x + h0.y * h0.y + h0.z * h0.z + h0.w * h0.w +
                      h1.x * h1.x + h1.y * h1.y + h1.z * h1.z + h1.w * h1.w;
#pragma unroll
            for (int off = 16; off > 0; off >>= 1) {
                s += __shfl_xor(s, off, 32);
                q += __shfl_xor(q, off, 32);
            }
            float mu = s * (1.0f / COUT);
            float var = q * (1.0f / COUT) - mu * mu;
            float rs = rsqrtf(var + LN_EPS);
            if (r < N) {
                int cb = lane * 8;
                float4 w0 = *(const float4*)(lnw + cb);
                float4 w1 = *(const float4*)(lnw + cb + 4);
                float4 b0 = *(const float4*)(lnb + cb);
                float4 b1 = *(const float4*)(lnb + cb + 4);
                float hv[8] = {h0.x, h0.y, h0.z, h0.w, h1.x, h1.y, h1.z, h1.w};
                float wv[8] = {w0.x, w0.y, w0.z, w0.w, w1.x, w1.y, w1.z, w1.w};
                float bv[8] = {b0.x, b0.y, b0.z, b0.w, b1.x, b1.y, b1.z, b1.w};
                float yv[8];
#pragma unroll
                for (int i = 0; i < 8; ++i) {
                    float y = (hv[i] - mu) * rs * wv[i] + bv[i];
                    yv[i] = (y >= 0.f) ? y : slope * y;
                }
                float* op = out + (long)r * COUT + cb;
                *(float4*)(op)     = make_float4(yv[0], yv[1], yv[2], yv[3]);
                *(float4*)(op + 4) = make_float4(yv[4], yv[5], yv[6], yv[7]);
            }
        }
        __syncthreads();
    }
}

extern "C" void kernel_launch(void* const* d_in, const int* in_sizes, int n_in,
                              void* d_out, int out_size, void* d_ws, size_t ws_size,
                              hipStream_t stream) {
    const float* x    = (const float*)d_in[0];
    const int*   bat  = (const int*)d_in[1];
    const float* gw   = (const float*)d_in[2];
    const float* gb   = (const float*)d_in[3];
    const float* lw   = (const float*)d_in[4];
    const float* lnw  = (const float*)d_in[5];
    const float* lnb  = (const float*)d_in[6];
    const float* pa   = (const float*)d_in[7];
    float* out = (float*)d_out;
    int N = in_sizes[0] / CIN;

    // workspace layout
    float*  xm  = (float*)d_ws;                       // [8192][128] f32 (4 MB)
    float*  lam = xm + (size_t)NSEG * CIN;            // [8192][256] f32 (8 MB)
    __bf16* gwh = (__bf16*)(lam + (size_t)NSEG * COUT);
    __bf16* gwl = gwh + COUT * CIN;
    __bf16* lwh = gwl + COUT * CIN;
    __bf16* lwl = lwh + COUT * CIN;

    fill_neginf<<<(NSEG * CIN + 255) / 256, 256, 0, stream>>>(xm, NSEG * CIN);
    cvt_weights<<<(COUT * CIN + 255) / 256, 256, 0, stream>>>(gw, lw, gwh, gwl, lwh, lwl);
    seg_max<<<(N + 255) / 256, 128, 0, stream>>>(x, bat, xm, N);
    lam_gemm<<<NSEG / TM, 512, 0, stream>>>(xm, lwh, lwl, lam);

    int nTiles = (N + TM - 1) / TM;
    int blocks = nTiles < 2048 ? nTiles : 2048;
    main_fused<<<blocks, 512, 0, stream>>>(x, bat, gwh, gwl, gb, lam, lnw, lnb, pa, out, N);
}
